// MambaBlock_27195732918562
// MI455X (gfx1250) — matrix operand order repaired and checked by reading, not compiled, not run on hardware
//
#include <hip/hip_runtime.h>
#include <hip/hip_bf16.h>

// ---------------------------------------------------------------------------
// Types for CDNA5 WMMA (wave32)
// ---------------------------------------------------------------------------
typedef __attribute__((ext_vector_type(16))) __bf16 v16bf;
typedef __attribute__((ext_vector_type(8)))  __bf16 v8bf;
typedef __attribute__((ext_vector_type(8)))  float  v8f;

#define D_MODEL 1024
#define N_HEADS 16
#define D_STATE 64

// LDS row strides (elements). 40*2B = 80B = 20 banks -> stride-20 mod 64 is
// conflict-free across 16 lanes, and rows stay 16B-aligned for b128 ops.
#define LDA 40
#define LDB 40

// ---------------------------------------------------------------------------
// Weight prep: fp32 K x N  ->  bf16 N x K (transposed), LDS-tiled 32x32 so
// both global read and global write stay coalesced. One-time cost (~10 MB).
// ---------------------------------------------------------------------------
__global__ __launch_bounds__(256) void transpose_to_bf16_kernel(
    const float* __restrict__ in,   // K x N fp32
    __bf16* __restrict__ out,       // N x K bf16
    int K, int N)
{
    __shared__ float sh[32][33];
    const int k0 = blockIdx.y * 32;
    const int n0 = blockIdx.x * 32;
    const int r  = threadIdx.x >> 3;        // 0..31
    const int c  = (threadIdx.x & 7) * 4;   // 0,4,...,28

    float4 v = *(const float4*)(in + (size_t)(k0 + r) * N + n0 + c);
    sh[r][c + 0] = v.x; sh[r][c + 1] = v.y;
    sh[r][c + 2] = v.z; sh[r][c + 3] = v.w;
    __syncthreads();

    __bf16* dst = out + (size_t)(n0 + r) * K + k0 + c;
    dst[0] = (__bf16)sh[c + 0][r];
    dst[1] = (__bf16)sh[c + 1][r];
    dst[2] = (__bf16)sh[c + 2][r];
    dst[3] = (__bf16)sh[c + 3][r];
}

// ---------------------------------------------------------------------------
// RMSNorm: one block per row of 1024. Writes xn (fp32) and xn (bf16).
// ---------------------------------------------------------------------------
__global__ __launch_bounds__(256) void rmsnorm_kernel(
    const float* __restrict__ x, const float* __restrict__ w,
    float* __restrict__ xn, __bf16* __restrict__ xnb)
{
    __shared__ float red[256];
    const int row = blockIdx.x;
    const int t   = threadIdx.x;
    const float* xr = x + (size_t)row * D_MODEL;

    float4 xv = *(const float4*)(xr + t * 4);
    float ss = xv.x * xv.x + xv.y * xv.y + xv.z * xv.z + xv.w * xv.w;
    red[t] = ss;
    __syncthreads();
    for (int off = 128; off > 0; off >>= 1) {
        if (t < off) red[t] += red[t + off];
        __syncthreads();
    }
    const float inv_rms = 1.0f / sqrtf(red[0] * (1.0f / (float)D_MODEL) + 1e-6f);

    float4 wv = *(const float4*)(w + t * 4);
    float4 o;
    o.x = wv.x * xv.x * inv_rms;
    o.y = wv.y * xv.y * inv_rms;
    o.z = wv.z * xv.z * inv_rms;
    o.w = wv.w * xv.w * inv_rms;

    float* xo = xn + (size_t)row * D_MODEL + t * 4;
    *(float4*)xo = o;
    __bf16* bo = xnb + (size_t)row * D_MODEL + t * 4;
    bo[0] = (__bf16)o.x; bo[1] = (__bf16)o.y;
    bo[2] = (__bf16)o.z; bo[3] = (__bf16)o.w;
}

// ---------------------------------------------------------------------------
// bf16 WMMA GEMM: out[M,N] = epilogue(A[M,K] @ W[K,N] + bias)
//   W is passed PRE-TRANSPOSED as Wt[N,K] so both LDS tiles fill with
//   ds_store_b128 (no scalar transpose in the hot loop).
//   mode 0: out = acc + bias
//   mode 1: out = sigmoid(acc + bias)
//   mode 2: out = acc + bias + resid
// Block: 256 threads (8 waves, 4x2), tile 128(M) x 128(N), K-step 32,
// double-buffered LDS (one barrier per K-step). Each wave owns a 32x64
// region = 2x4 v_wmma_f32_16x16x32_bf16 tiles; the 4 B fragments are loaded
// once per K-step and reused across both A rows (1.5 ds_load per WMMA).
// ---------------------------------------------------------------------------
__global__ __launch_bounds__(256) void gemm_bf16_wmma_kernel(
    const __bf16* __restrict__ A,   // M x K row-major
    const __bf16* __restrict__ Wt,  // N x K row-major (pre-transposed)
    const float* __restrict__ bias, const float* __restrict__ resid,
    float* __restrict__ out, int M, int N, int K, int mode)
{
    __shared__ __bf16 sA[2][128 * LDA];   // [m][k]
    __shared__ __bf16 sB[2][128 * LDB];   // [n][k]

    const int tid   = threadIdx.x;
    const int lane  = tid & 31;
    const int wave  = tid >> 5;    // 0..7
    const int wm    = wave >> 1;   // 0..3  (M direction)
    const int wn    = wave & 1;    // 0..1  (N direction)
    const int khalf = lane >> 4;   // 0/1 : low/high K half in this kernel's k order
    const int l16   = lane & 15;

    const int m0 = blockIdx.y * 128;
    const int n0 = blockIdx.x * 128;

    // Tile-fill mappings: both tiles are 128 rows x 32 k, 16 elems/thread
    const int f_row = tid >> 1;            // 0..127
    const int f_col = (tid & 1) * 16;      // 0 or 16

    const __bf16* aSrc = A  + (size_t)(m0 + f_row) * K + f_col;
    const __bf16* bSrc = Wt + (size_t)(n0 + f_row) * K + f_col;

    uint4 ra0, ra1, rb0, rb1;
    auto gload = [&](int kk) {
        ra0 = *(const uint4*)(aSrc + kk);
        ra1 = *(const uint4*)(aSrc + kk + 8);
        rb0 = *(const uint4*)(bSrc + kk);
        rb1 = *(const uint4*)(bSrc + kk + 8);
    };
    auto sstore = [&](int buf) {
        *(uint4*)(&sA[buf][f_row * LDA + f_col])     = ra0;
        *(uint4*)(&sA[buf][f_row * LDA + f_col + 8]) = ra1;
        *(uint4*)(&sB[buf][f_row * LDB + f_col])     = rb0;
        *(uint4*)(&sB[buf][f_row * LDB + f_col + 8]) = rb1;
    };

    const v8f vzero = {0.f,0.f,0.f,0.f,0.f,0.f,0.f,0.f};
    v8f acc[2][4];
    #pragma unroll
    for (int ti = 0; ti < 2; ++ti)
        #pragma unroll
        for (int tj = 0; tj < 4; ++tj)
            acc[ti][tj] = vzero;

    gload(0);
    sstore(0);
    __syncthreads();

    int cur = 0;
    for (int kk = 32; kk <= K; kk += 32) {
        const bool more = (kk < K);
        if (more) gload(kk);   // prefetch next tile while WMMAs run

        // B fragments 32x16 (lane col = n, K = 16*khalf..16*khalf+15),
        // loaded once and reused for both A-row tiles.
        v16bf bfrag[4];
        #pragma unroll
        for (int tj = 0; tj < 4; ++tj) {
            const int ncol = wn * 64 + tj * 16 + l16;
            v8bf blo = *(const v8bf*)(&sB[cur][ncol * LDB + khalf * 16]);
            v8bf bhi = *(const v8bf*)(&sB[cur][ncol * LDB + khalf * 16 + 8]);
            bfrag[tj] = __builtin_shufflevector(blo, bhi,
                0,1,2,3,4,5,6,7,8,9,10,11,12,13,14,15);
        }

        #pragma unroll
        for (int ti = 0; ti < 2; ++ti) {
            // A fragment 16x32: lane row = m, frag[0..15]=K(16*khalf..16*khalf+15),
            // matching the B fragment K order.
            const int mrow = wm * 32 + ti * 16 + l16;
            v8bf alo = *(const v8bf*)(&sA[cur][mrow * LDA + khalf * 16]);
            v8bf ahi = *(const v8bf*)(&sA[cur][mrow * LDA + khalf * 16 + 8]);
            v16bf afrag = __builtin_shufflevector(alo, ahi,
                0,1,2,3,4,5,6,7,8,9,10,11,12,13,14,15);
            #pragma unroll
            for (int tj = 0; tj < 4; ++tj) {
                acc[ti][tj] = __builtin_amdgcn_wmma_f32_16x16x32_bf16(
                    false, afrag, false, bfrag[tj], (short)0, acc[ti][tj],
                    false, false);
            }
        }

        if (more) {
            sstore(cur ^ 1);   // previous barrier guarantees buf^1 is free
            __syncthreads();
        }
        cur ^= 1;
    }

    // Epilogue: C/D layout -> VGPR r of lane L is (row = r + 8*khalf, col = l16)
    #pragma unroll
    for (int ti = 0; ti < 2; ++ti) {
        #pragma unroll
        for (int tj = 0; tj < 4; ++tj) {
            const int col  = n0 + wn * 64 + tj * 16 + l16;
            const float bc = bias[col];
            #pragma unroll
            for (int r = 0; r < 8; ++r) {
                const int row = m0 + wm * 32 + ti * 16 + r + 8 * khalf;
                const size_t idx = (size_t)row * N + col;
                float v = acc[ti][tj][r] + bc;
                if (mode == 1)      v = 1.0f / (1.0f + __expf(-v));
                else if (mode == 2) v += resid[idx];
                out[idx] = v;
            }
        }
    }
}

// ---------------------------------------------------------------------------
// Serial SSM scan: one thread per (b, head, channel) = 4096 threads.
// h_s = sigmoid(delta)*h_{s-1} + B;  ssm = C*h;
// mixed = gate*ssm + (1-gate)*xn  (stored bf16 for the final GEMM)
// ---------------------------------------------------------------------------
__global__ __launch_bounds__(256) void scan_kernel(
    const float* __restrict__ p,      // [B*S, 3072] = (head, {delta,B,C}, 64)
    const float* __restrict__ gate,   // [B*S, 1024]
    const float* __restrict__ xn,     // [B*S, 1024]
    const float* __restrict__ state0, // [B, 16, 64]
    __bf16* __restrict__ mixed,       // [B*S, 1024]
    float* __restrict__ state_out,    // [B, 16, 64]
    int S)
{
    const int t   = blockIdx.x * 256 + threadIdx.x;  // 0..B*1024-1
    const int b   = t >> 10;
    const int rem = t & 1023;        // model-dim index = head*64 + ch
    const int hh  = rem >> 6;
    const int ch  = rem & 63;

    float h = state0[t];
    size_t pbase = (size_t)b * S * 3072 + hh * 192 + ch;
    size_t gbase = (size_t)b * S * 1024 + rem;

    for (int s = 0; s < S; ++s) {
        const float dpre = p[pbase];
        const float bt   = p[pbase + 64];
        const float ct   = p[pbase + 128];
        const float g    = gate[gbase];
        const float xv   = xn[gbase];

        const float d = 1.0f / (1.0f + __expf(-dpre));
        h = d * h + bt;
        const float mix = g * (ct * h) + (1.0f - g) * xv;
        mixed[gbase] = (__bf16)mix;

        pbase += 3072;
        gbase += 1024;
    }
    state_out[t] = h;
}

// ---------------------------------------------------------------------------
// Host orchestration
// ---------------------------------------------------------------------------
extern "C" void kernel_launch(void* const* d_in, const int* in_sizes, int n_in,
                              void* d_out, int out_size, void* d_ws, size_t ws_size,
                              hipStream_t stream)
{
    const float* x      = (const float*)d_in[0];   // (B,S,1024)
    const float* state0 = (const float*)d_in[1];   // (B,16,64)
    const float* norm_w = (const float*)d_in[2];   // (1024,)
    const float* Wp     = (const float*)d_in[3];   // (1024,3072)
    const float* bp     = (const float*)d_in[4];   // (3072,)
    const float* Wg     = (const float*)d_in[5];   // (1024,1024)
    const float* bg     = (const float*)d_in[6];   // (1024,)
    const float* Wo     = (const float*)d_in[7];   // (1024,1024)
    const float* bo     = (const float*)d_in[8];   // (1024,)

    const int B = in_sizes[1] / (N_HEADS * D_STATE);      // 4
    const int M = in_sizes[0] / D_MODEL;                  // B*S = 16384
    const int S = M / B;                                  // 4096

    // ---- workspace layout (256B-aligned slabs) ----
    char* ws = (char*)d_ws;
    size_t off = 0;
    auto alloc = [&](size_t bytes) {
        void* p = ws + off;
        off += (bytes + 255) & ~(size_t)255;
        return p;
    };
    float*  xn_f32   = (float*) alloc((size_t)M * D_MODEL * 4);
    __bf16* xn_bf    = (__bf16*)alloc((size_t)M * D_MODEL * 2);
    float*  p_f32    = (float*) alloc((size_t)M * 3072 * 4);
    __bf16* mixed_bf = (__bf16*)alloc((size_t)M * D_MODEL * 2);
    __bf16* WgT_bf   = (__bf16*)alloc((size_t)D_MODEL * 1024 * 2);  // N x K
    __bf16* WpT_bf   = (__bf16*)alloc((size_t)D_MODEL * 3072 * 2);  // N x K
    __bf16* WoT_bf   = (__bf16*)alloc((size_t)D_MODEL * 1024 * 2);  // N x K
    (void)ws_size; (void)n_in; (void)out_size;

    // gate lives in d_out[0 : M*1024): consumed by scan_kernel, then fully
    // overwritten by the final GEMM.
    float* gate      = (float*)d_out;
    float* y_out     = (float*)d_out;
    float* state_out = (float*)d_out + (size_t)M * D_MODEL;

    // 1) weights fp32 K x N -> bf16 N x K (transposed for the GEMM B-tiles)
    {
        dim3 g1(1024 / 32, D_MODEL / 32);
        transpose_to_bf16_kernel<<<g1, 256, 0, stream>>>(Wg, WgT_bf, D_MODEL, 1024);
        dim3 g2(3072 / 32, D_MODEL / 32);
        transpose_to_bf16_kernel<<<g2, 256, 0, stream>>>(Wp, WpT_bf, D_MODEL, 3072);
        dim3 g3(1024 / 32, D_MODEL / 32);
        transpose_to_bf16_kernel<<<g3, 256, 0, stream>>>(Wo, WoT_bf, D_MODEL, 1024);
    }

    // 2) RMSNorm
    rmsnorm_kernel<<<M, 256, 0, stream>>>(x, norm_w, xn_f32, xn_bf);

    // 3) gate = sigmoid(xn @ Wg + bg)
    {
        dim3 grid(1024 / 128, M / 128);
        gemm_bf16_wmma_kernel<<<grid, 256, 0, stream>>>(
            xn_bf, WgT_bf, bg, nullptr, gate, M, 1024, D_MODEL, /*mode=*/1);
    }

    // 4) p = xn @ Wp + bp
    {
        dim3 grid(3072 / 128, M / 128);
        gemm_bf16_wmma_kernel<<<grid, 256, 0, stream>>>(
            xn_bf, WpT_bf, bp, nullptr, p_f32, M, 3072, D_MODEL, /*mode=*/0);
    }

    // 5) serial scan + gating blend
    scan_kernel<<<(B * D_MODEL) / 256, 256, 0, stream>>>(
        p_f32, gate, xn_f32, state0, mixed_bf, state_out, S);

    // 6) y = mixed @ Wo + bo + x
    {
        dim3 grid(1024 / 128, M / 128);
        gemm_bf16_wmma_kernel<<<grid, 256, 0, stream>>>(
            mixed_bf, WoT_bf, bo, x, y_out, M, 1024, D_MODEL, /*mode=*/2);
    }
}
